// MultiHeadAttention_20770461843539
// MI455X (gfx1250) — matrix operand order repaired and checked by reading, not compiled, not run on hardware
//
#include <hip/hip_runtime.h>
#include <hip/hip_fp16.h>

// ---------------------------------------------------------------------------
// MHA w/ relative position bias on gfx1250 (MI455X), wave32 + WMMA f16->f32,
// TDM (tensor_load_to_lds) tile staging.
//   B=128, N=256, D_MODEL=512, H=8, D_H=64, MAX_LEN=256
// Pipeline:
//   0) cvt kernels       : query/key/value -> f16; Wq/Wk/Wv/Wo -> f16 transposed
//   1) gemm<F16>  (TDM)  : Q = query*Wq (scale 1/8 folded), f16 out
//   2) gemm<F16>  (TDM)  : K = key*Wk
//   3) gemm<F16T> (TDM)  : Vt = (value*Wv)^T -> [e][m]
//   4) attn kernel       : softmax(Q K^T + bias) V -> concat f16
//   5) gemm<F32>  (TDM)  : out = concat * Wo
// ---------------------------------------------------------------------------

typedef __attribute__((ext_vector_type(16))) _Float16 v16h;
typedef __attribute__((ext_vector_type(8)))  float    v8f;
typedef __attribute__((ext_vector_type(4)))  unsigned v4u;
typedef __attribute__((ext_vector_type(4)))  unsigned u32x4;
typedef __attribute__((ext_vector_type(8)))  int      i32x8;
typedef __attribute__((ext_vector_type(4)))  int      i32x4;

union Frag16 { v16h v; _Float16 h[16]; v4u q[2]; unsigned u[8]; };

__device__ __forceinline__ v8f wmma_f16f32(v16h a, v16h b, v8f c) {
  return __builtin_amdgcn_wmma_f32_16x16x32_f16(false, a, false, b, (short)0, c,
                                                false, false);
}

// ---------------------------------------------------------------------------
// TDM: 2D tile (tile_rows x tile_k f16) from global (row pitch in elements)
// into LDS at lds_addr with padded rows: pad_interval=16 dwords (64B = 32
// halves), pad_amount=4 dwords (8 halves) -> LDS row stride 40 halves.
// D# layout per CDNA5 ISA (groups 0/1; groups 2/3 zero).
// amdgpu-toolchain (clang-23) 6-arg builtin form.
// ---------------------------------------------------------------------------
__device__ __forceinline__ void tdm_load_2d(unsigned lds_addr,
                                            unsigned long long gaddr,
                                            unsigned tile_k, unsigned tile_rows,
                                            unsigned row_pitch_elems) {
  u32x4 g0;
  g0[0] = 1u;                                        // count=1 (valid user D#)
  g0[1] = lds_addr;                                  // LDS byte address
  g0[2] = (unsigned)(gaddr & 0xFFFFFFFFull);         // global_addr[31:0]
  g0[3] = (unsigned)((gaddr >> 32) & 0x1FFFFFFull)   // global_addr[56:32]
          | (2u << 30);                               // type=2 ("image")
  i32x8 g1;
  g1[0] = (int)((1u << 16)        // data_size = 1 -> 2 bytes
                | (1u << 20)      // pad_enable
                | (3u << 22)      // pad_interval: 16 dwords (64B) per row
                | (3u << 25));    // pad_amount: 4 dwords (16B)
  g1[1] = (int)((tile_k & 0xFFFFu) << 16);                    // tensor_dim0 lo
  g1[2] = (int)((tile_k >> 16) | ((tile_rows & 0xFFFFu) << 16)); // td0hi|td1lo
  g1[3] = (int)((tile_rows >> 16) | ((tile_k & 0xFFFFu) << 16)); // td1hi|tile0
  g1[4] = (int)(tile_rows & 0xFFFFu);                 // tile_dim1; tile_dim2=0
  g1[5] = (int)row_pitch_elems;                       // tensor_dim0_stride lo
  g1[6] = 0;                                          // stride0 hi | stride1 lo
  g1[7] = 0;
  i32x4 z4 = {0, 0, 0, 0};
  i32x8 z8 = {0, 0, 0, 0, 0, 0, 0, 0};
  __builtin_amdgcn_tensor_load_to_lds(g0, g1, z4, z4, z8, 0);
}

// A fragment (16x32 f16, MxK): lane<16 holds M=lane,
// halves {0..7,8..15} -> K {kb+kg..+7, kb+16+kg..+7}, kg=8*(lane>=16)
__device__ __forceinline__ v16h ld_afrag_g(const _Float16* __restrict__ base,
                                           int ld, int m0, int kb, int lane) {
  int row = m0 + (lane & 15);
  int kg  = (lane >> 4) << 3;
  const _Float16* p = base + (long long)row * ld + kb + kg;
  Frag16 f;
  f.q[0] = *(const v4u*)p;
  f.q[1] = *(const v4u*)(p + 16);
  return f.v;
}

// B fragment (32x16 f16, KxN) from transposed storage: (k,n) at baseT[n*ld+k].
// Lane holds N = n0+(lane&15), K {kb+kg..+7, kb+16+kg..+7}, kg=8*(lane>=16).
__device__ __forceinline__ v16h ld_bfrag_g(const _Float16* __restrict__ baseT,
                                           int ld, int n0, int kb, int lane) {
  int n = n0 + (lane & 15);
  int k = kb + ((lane >> 4) << 3);
  const _Float16* p = baseT + (long long)n * ld + k;
  Frag16 f;
  f.q[0] = *(const v4u*)p;
  f.q[1] = *(const v4u*)(p + 16);
  return f.v;
}

// ---------------------------------------------------------------------------
// Conversion kernels (run once; removes all per-GEMM-block f32->f16 work)
// ---------------------------------------------------------------------------
__global__ __launch_bounds__(256) void cvt_f16(const float* __restrict__ s,
                                               _Float16* __restrict__ d,
                                               int n8) {
  int i = blockIdx.x * 256 + threadIdx.x;
  if (i >= n8) return;
  long long base = (long long)i * 8;
  union { v4u q; _Float16 h[8]; } pk;
#pragma unroll
  for (int j = 0; j < 8; ++j) pk.h[j] = (_Float16)s[base + j];
  *(v4u*)(d + base) = pk.q;
}

// WT[batch][e][k] = (f16) W[batch][k][e]
__global__ __launch_bounds__(256) void cvt_wT(const float* __restrict__ W,
                                              _Float16* __restrict__ WT,
                                              int kdim, int edim) {
  long long b = (long long)blockIdx.y * kdim * edim;
  int idx = blockIdx.x * 256 + threadIdx.x;
  if (idx >= kdim * edim) return;
  int kk = idx / edim, ee = idx % edim;
  WT[b + (long long)ee * kdim + kk] = (_Float16)W[b + idx];
}

// ---------------------------------------------------------------------------
// Batched GEMM with TDM staging: C[batch] (256 x 64-tile) = A (256xK) * BT^T
// A f16 row-major (lda), BT f16 transposed [n][k] (ldbT). Block = 256 thr
// (8 waves); wave tile 32x64; K-step 32. Each wave TDM-loads its own 32-row
// A slice and 8-row B slice (EXEC ignored by TDM; all 8 descriptors useful).
// ---------------------------------------------------------------------------
enum { OUT_F16 = 0, OUT_F16T = 1, OUT_F32 = 2 };

template <int OUT>
__global__ __launch_bounds__(256) void gemm_tdm(
    const _Float16* __restrict__ A, const _Float16* __restrict__ BT,
    void* __restrict__ Cv,
    int lda, int ldbT, int ldc, int kdim, int a_shr, int b_mask,
    long long a_stride, long long b_stride, long long c_stride, float scale) {
  __shared__ _Float16 lds_a[256][40];   // padded stride 40 (TDM pad)
  __shared__ _Float16 lds_b[64][40];    // transposed W tile [n][k]

  const int t     = threadIdx.x;
  const int lane  = t & 31;
  const int batch = blockIdx.y;
  const int n0    = blockIdx.x * 64;
  const int rw    = __builtin_amdgcn_readfirstlane((t >> 5) * 32);  // wave row
  const int wb    = rw >> 2;                                        // wave*8

  const _Float16* Ab = A  + (long long)(batch >> a_shr) * a_stride;
  const _Float16* Bb = BT + (long long)(batch & b_mask) * b_stride;

  const unsigned lab = (unsigned)(uintptr_t)&lds_a[0][0];
  const unsigned lbb = (unsigned)(uintptr_t)&lds_b[0][0];

  v8f zero = {};
  v8f acc[2][4];
#pragma unroll
  for (int i = 0; i < 2; ++i)
#pragma unroll
    for (int j = 0; j < 4; ++j) acc[i][j] = zero;

  for (int kb = 0; kb < kdim; kb += 32) {
    // async 2D copies: this wave's A rows [rw, rw+32), B rows [wb, wb+8)
    tdm_load_2d(lab + (unsigned)rw * 80,
                (unsigned long long)(uintptr_t)(Ab + (long long)rw * lda + kb),
                32u, 32u, (unsigned)lda);
    tdm_load_2d(lbb + (unsigned)wb * 80,
                (unsigned long long)(uintptr_t)(Bb +
                    (long long)(n0 + wb) * ldbT + kb),
                32u, 8u, (unsigned)ldbT);
    __builtin_amdgcn_s_wait_tensorcnt(0);
    __syncthreads();

    Frag16 a0, a1;
    {
      int row = rw + (lane & 15);
      int kg  = (lane >> 4) << 3;
      const _Float16* p0 = &lds_a[row][kg];
      const _Float16* p1 = &lds_a[row + 16][kg];
      a0.q[0] = *(const v4u*)p0; a0.q[1] = *(const v4u*)(p0 + 16);
      a1.q[0] = *(const v4u*)p1; a1.q[1] = *(const v4u*)(p1 + 16);
    }
#pragma unroll
    for (int ct = 0; ct < 4; ++ct) {
      Frag16 bf;
      int n = ct * 16 + (lane & 15);
      int k = (lane >> 4) << 3;
      const _Float16* p = &lds_b[n][k];
      bf.q[0] = *(const v4u*)p; bf.q[1] = *(const v4u*)(p + 16);
      acc[0][ct] = wmma_f16f32(a0.v, bf.v, acc[0][ct]);
      acc[1][ct] = wmma_f16f32(a1.v, bf.v, acc[1][ct]);
    }
    __syncthreads();   // protect LDS before next iteration's TDM overwrite
  }

  // epilogue: C layout = VGPR r -> M = r + 8*(lane>=16), N = lane&15
#pragma unroll
  for (int rt = 0; rt < 2; ++rt) {
    int rb = rw + rt * 16 + ((lane >> 4) << 3);
#pragma unroll
    for (int ct = 0; ct < 4; ++ct) {
      int col = n0 + ct * 16 + (lane & 15);
      v8f c = acc[rt][ct];
      if (OUT == OUT_F32) {
        float* C = (float*)Cv + (long long)batch * c_stride;
#pragma unroll
        for (int r = 0; r < 8; ++r)
          C[(long long)(rb + r) * ldc + col] = c[r] * scale;
      } else if (OUT == OUT_F16) {
        _Float16* C = (_Float16*)Cv + (long long)batch * c_stride;
#pragma unroll
        for (int r = 0; r < 8; ++r)
          C[(long long)(rb + r) * ldc + col] = (_Float16)(c[r] * scale);
      } else { // OUT_F16T: Ct[col][row], 8 contiguous rows -> 16B store
        _Float16* C = (_Float16*)Cv + (long long)batch * c_stride;
        union { v4u q; _Float16 h[8]; } pk;
#pragma unroll
        for (int r = 0; r < 8; ++r) pk.h[r] = (_Float16)(c[r] * scale);
        *(v4u*)&C[(long long)col * ldc + rb] = pk.q;
      }
    }
  }
}

// ---------------------------------------------------------------------------
// Attention: one block = 128 query rows of one (b,h). 128 thr = 4 waves.
//   A: S = Q K^T -> LDS f32 [128][257] (Q resident in VGPRs)
//   B: per-row bias + softmax; row rewritten in place as unnormalized f16
//   C: O = E * V via WMMA; scale by 1/sum; store concat f16
// ---------------------------------------------------------------------------
#define SROW 257

__device__ __forceinline__ v16h ld_afrag_E(const char* sbase, int m0, int kb,
                                           int lane) {
  int row = m0 + (lane & 15);
  int kg  = (lane >> 4) << 3;
  const char* p = sbase + row * (SROW * 4) + 2 * (kb + kg);
  Frag16 f;
#pragma unroll
  for (int i = 0; i < 4; ++i) f.u[i]     = *(const unsigned*)(p + 4 * i);
#pragma unroll
  for (int i = 0; i < 4; ++i) f.u[4 + i] = *(const unsigned*)(p + 32 + 4 * i);
  return f.v;
}

__global__ __launch_bounds__(128) void attn_kernel(
    const _Float16* __restrict__ Q, const _Float16* __restrict__ K,
    const _Float16* __restrict__ Vt, const float* __restrict__ rpe,
    _Float16* __restrict__ concat) {
  extern __shared__ char smem[];
  float* S    = (float*)smem;                       // [128][SROW]
  float* bias = (float*)(smem + 128 * SROW * 4);    // [513], prescaled by 1/8
  float* rsum = bias + 513;                         // [128]

  const int t    = threadIdx.x;
  const int lane = t & 31;
  const int wv   = t >> 5;
  const int half = blockIdx.x;
  const int bh   = blockIdx.y;
  const int h    = bh & 7;
  const int b    = bh >> 3;

  const _Float16* Qb = Q  + (long long)bh * (256 * 64);
  const _Float16* Kb = K  + (long long)bh * (256 * 64);
  const _Float16* Vb = Vt + (long long)bh * (64 * 256);

  for (int j = t; j < 513; j += 128) bias[j] = rpe[h * 513 + j] * 0.125f;

  const int mloc = wv * 32;
  const int mg   = half * 128 + mloc;

  v16h q00 = ld_afrag_g(Qb, 64, mg,      0,  lane);
  v16h q01 = ld_afrag_g(Qb, 64, mg,      32, lane);
  v16h q10 = ld_afrag_g(Qb, 64, mg + 16, 0,  lane);
  v16h q11 = ld_afrag_g(Qb, 64, mg + 16, 32, lane);

#pragma unroll 4
  for (int ct = 0; ct < 16; ++ct) {
    v16h b0 = ld_bfrag_g(Kb, 64, ct * 16, 0,  lane);
    v16h b1 = ld_bfrag_g(Kb, 64, ct * 16, 32, lane);
    v8f c0 = {}; v8f c1 = {};
    c0 = wmma_f16f32(q00, b0, c0); c0 = wmma_f16f32(q01, b1, c0);
    c1 = wmma_f16f32(q10, b0, c1); c1 = wmma_f16f32(q11, b1, c1);
    int col = ct * 16 + (lane & 15);
    int rb0 = mloc + ((lane >> 4) << 3);
#pragma unroll
    for (int r = 0; r < 8; ++r) S[(rb0 + r) * SROW + col]      = c0[r];
#pragma unroll
    for (int r = 0; r < 8; ++r) S[(rb0 + 16 + r) * SROW + col] = c1[r];
  }
  __syncthreads();

  { // softmax, one thread per row; in-place f32 -> f16 (2j trails 4j)
    int row = t;
    int mq  = half * 128 + row;
    float* sr = S + row * SROW;
    float mx = -3.0e38f;
#pragma unroll 8
    for (int j = 0; j < 256; ++j) {
      float v = sr[j] + bias[mq - j + 256];
      sr[j] = v;
      mx = fmaxf(mx, v);
    }
    _Float16* er = (_Float16*)sr;
    float sum = 0.0f;
#pragma unroll 8
    for (int j = 0; j < 256; ++j) {
      float e = __expf(sr[j] - mx);
      sum += e;
      er[j] = (_Float16)e;
    }
    rsum[row] = 1.0f / sum;
  }
  __syncthreads();

  v8f zero = {};
  v8f o[2][4];
#pragma unroll
  for (int i = 0; i < 2; ++i)
#pragma unroll
    for (int j = 0; j < 4; ++j) o[i][j] = zero;

#pragma unroll 2
  for (int kb = 0; kb < 256; kb += 32) {
    v16h a0 = ld_afrag_E(smem, mloc,      kb, lane);
    v16h a1 = ld_afrag_E(smem, mloc + 16, kb, lane);
#pragma unroll
    for (int ct = 0; ct < 4; ++ct) {
      v16h bf = ld_bfrag_g(Vb, 256, ct * 16, kb, lane);
      o[0][ct] = wmma_f16f32(a0, bf, o[0][ct]);
      o[1][ct] = wmma_f16f32(a1, bf, o[1][ct]);
    }
  }

#pragma unroll
  for (int rt = 0; rt < 2; ++rt) {
    int rb = mloc + rt * 16 + ((lane >> 4) << 3);
#pragma unroll
    for (int ct = 0; ct < 4; ++ct) {
      int e = ct * 16 + (lane & 15);
#pragma unroll
      for (int r = 0; r < 8; ++r) {
        int rl = rb + r;
        float val = o[rt][ct][r] * rsum[rl];
        int mq = half * 128 + rl;
        concat[((long long)(b * 256 + mq)) * 512 + h * 64 + e] = (_Float16)val;
      }
    }
  }
}

// ---------------------------------------------------------------------------
extern "C" void kernel_launch(void* const* d_in, const int* in_sizes, int n_in,
                              void* d_out, int out_size, void* d_ws,
                              size_t ws_size, hipStream_t stream) {
  const float* key   = (const float*)d_in[0];
  const float* query = (const float*)d_in[1];
  const float* value = (const float*)d_in[2];
  const float* Wq    = (const float*)d_in[3];
  const float* Wk    = (const float*)d_in[4];
  const float* Wv    = (const float*)d_in[5];
  const float* Wo    = (const float*)d_in[6];
  const float* rpe   = (const float*)d_in[7];
  float* out = (float*)d_out;

  const size_t NB = (size_t)128 * 256 * 512;   // elements of one big tensor
  _Float16* q16    = (_Float16*)d_ws;          // converted activations (f16)
  _Float16* k16    = q16  + NB;
  _Float16* v16    = k16  + NB;
  _Float16* WqT    = v16  + NB;                // 8*64*512
  _Float16* WkT    = WqT  + (size_t)8 * 64 * 512;
  _Float16* WvT    = WkT  + (size_t)8 * 64 * 512;
  _Float16* WoT    = WvT  + (size_t)8 * 64 * 512;   // 512*512
  _Float16* Qb     = WoT  + (size_t)512 * 512;      // 1024*256*64 each
  _Float16* Kb     = Qb   + NB;
  _Float16* Vtb    = Kb   + NB;
  _Float16* concat = Vtb  + NB;

  // 0) conversions
  {
    int n8 = (int)(NB / 8);
    dim3 cb(256), cg((n8 + 255) / 256);
    cvt_f16<<<cg, cb, 0, stream>>>(query, q16, n8);
    cvt_f16<<<cg, cb, 0, stream>>>(key,   k16, n8);
    cvt_f16<<<cg, cb, 0, stream>>>(value, v16, n8);
    cvt_wT<<<dim3(128, 8), cb, 0, stream>>>(Wq, WqT, 512, 64);
    cvt_wT<<<dim3(128, 8), cb, 0, stream>>>(Wk, WkT, 512, 64);
    cvt_wT<<<dim3(128, 8), cb, 0, stream>>>(Wv, WvT, 512, 64);
    cvt_wT<<<dim3(1024, 1), cb, 0, stream>>>(Wo, WoT, 512, 512);
  }

  dim3 blk(256);
  dim3 g1(1, 1024);
  const long long sA = 256LL * 512, sB = 64LL * 512, sC = 256LL * 64;

  // 1-3) projections (Q carries 1/sqrt(64) = 1/8)
  gemm_tdm<OUT_F16><<<g1, blk, 0, stream>>>(
      q16, WqT, Qb, 512, 512, 64, 512, 3, 7, sA, sB, sC, 0.125f);
  gemm_tdm<OUT_F16><<<g1, blk, 0, stream>>>(
      k16, WkT, Kb, 512, 512, 64, 512, 3, 7, sA, sB, sC, 1.0f);
  gemm_tdm<OUT_F16T><<<g1, blk, 0, stream>>>(
      v16, WvT, Vtb, 512, 512, 256, 512, 3, 7, sA, sB, sC, 1.0f);

  // 4) attention (~131 KB dynamic LDS; 320 KB/WGP on CDNA5)
  int smem = 128 * SROW * 4 + 513 * 4 + 128 * 4;
  (void)hipFuncSetAttribute((const void*)attn_kernel,
                            hipFuncAttributeMaxDynamicSharedMemorySize, smem);
  attn_kernel<<<dim3(2, 1024), dim3(128), smem, stream>>>(Qb, Kb, Vtb, rpe,
                                                          concat);

  // 5) out = concat * Wo
  gemm_tdm<OUT_F32><<<dim3(8, 128), blk, 0, stream>>>(
      concat, WoT, out, 512, 512, 512, 512, 0, 0, 256LL * 512, 0LL,
      256LL * 512, 1.0f);
}
